// SSD_79302276153636
// MI455X (gfx1250) — matrix operand order repaired and checked by reading, not compiled, not run on hardware
//
#include <hip/hip_runtime.h>
#include <hip/hip_bf16.h>
#include <math.h>

// ---------------------------------------------------------------------------
// Types for CDNA5 WMMA / TDM
// ---------------------------------------------------------------------------
typedef __attribute__((ext_vector_type(16))) __bf16 v16bf;
typedef __attribute__((ext_vector_type(8)))  float  v8f;
typedef __attribute__((ext_vector_type(4)))  unsigned int u32x4;
typedef __attribute__((ext_vector_type(8)))  int i32x8;
typedef __attribute__((ext_vector_type(4)))  int i32x4;

struct Frag32B { uint4 lo, hi; };   // 32 bytes == v16bf

// ---------------------------------------------------------------------------
// bf16 helpers (storage = unsigned short, RNE conversion)
// ---------------------------------------------------------------------------
__device__ __forceinline__ float bf2f(unsigned short h) {
    unsigned u = ((unsigned)h) << 16;
    return __builtin_bit_cast(float, u);
}
__device__ __forceinline__ unsigned short f2bf(float f) {
    unsigned u = __builtin_bit_cast(unsigned, f);
    if ((u & 0x7F800000u) == 0x7F800000u) return (unsigned short)(u >> 16); // inf/nan
    unsigned r = u + 0x7FFFu + ((u >> 16) & 1u);
    return (unsigned short)(r >> 16);
}

__global__ void f32_to_bf16_k(const float* __restrict__ src,
                              unsigned short* __restrict__ dst, int n) {
    int i = blockIdx.x * blockDim.x + threadIdx.x;
    if (i < n) dst[i] = f2bf(src[i]);
}

// Weight convert + pad: [Cout,Ktot] f32 -> [Coutp,Ktotp] bf16 (zero padded)
__global__ void w_cvt_pad_k(const float* __restrict__ src,
                            unsigned short* __restrict__ dst,
                            int Cout, int Ktot, int Coutp, int Ktotp) {
    int i = blockIdx.x * blockDim.x + threadIdx.x;
    int tot = Coutp * Ktotp;
    if (i >= tot) return;
    int k = i % Ktotp, m = i / Ktotp;
    dst[i] = (m < Cout && k < Ktot) ? f2bf(src[m * Ktot + k]) : (unsigned short)0;
}

// ---------------------------------------------------------------------------
// TDM: 2D tile load Global -> LDS (weights tile, 64 rows x 32 bf16 elements,
// LDS row stride padded 64B + 16B). Descriptor per CDNA5 ISA D# layout.
// ---------------------------------------------------------------------------
__device__ __forceinline__ void tdm_load_a_tile(unsigned lds_addr,
                                                const unsigned short* gptr,
                                                int remK_elems, int remM_rows,
                                                int strideK_elems)
{
    unsigned long long ga = (unsigned long long)gptr;
    u32x4 g0;
    g0[0] = 1u;                                     // count=1, user descriptor
    g0[1] = lds_addr;                               // lds_addr [63:32]
    g0[2] = (unsigned)(ga & 0xFFFFFFFFu);           // global_addr lo
    g0[3] = (unsigned)((ga >> 32) & 0x1FFFFFFu)     // global_addr [56:32]
          | (2u << 30);                             // type=2 (image)
    i32x8 g1;
    g1[0] = (1 << 16)      // data_size = 2 bytes
          | (1 << 20)      // pad_enable
          | (3 << 22)      // pad_interval: 16 dwords (64B = one tile row)
          | (3 << 25);     // pad_amount: 4 dwords (16B)
    g1[1] = (remK_elems & 0xFFFF) << 16;                          // dim0 lo16
    g1[2] = ((unsigned)remK_elems >> 16) | ((remM_rows & 0xFFFF) << 16); // dim0 hi / dim1 lo
    g1[3] = ((unsigned)remM_rows >> 16) | (32 << 16);             // dim1 hi / tile_dim0=32
    g1[4] = 64;                                                   // tile_dim1=64, tile_dim2=0
    g1[5] = strideK_elems;                                        // dim0_stride lo32
    g1[6] = 0;                                                    // stride hi, dim1_stride lo
    g1[7] = 0;
    i32x4 z4 = {0, 0, 0, 0};
#if __clang_major__ >= 23
    i32x8 z8 = {0, 0, 0, 0, 0, 0, 0, 0};
    __builtin_amdgcn_tensor_load_to_lds(g0, g1, z4, z4, z8, 0);
#else
    __builtin_amdgcn_tensor_load_to_lds(g0, g1, z4, z4, 0);
#endif
}

// ---------------------------------------------------------------------------
// Implicit-GEMM convolution via v_wmma_f32_16x16x32_bf16.
// GEMM: M = Cout (padded to 64), N = B*Ho*Wo, K = Cin*kh*kw (padded to 32).
// Block: 256 threads / 8 waves; tile 64(M) x 128(N) x 32(K).
// Waves arranged 2(M) x 4(N); each wave computes 2x2 16x16 sub-tiles (4 WMMA).
// A tile staged by the Tensor Data Mover; B tile = im2col by all threads.
// ---------------------------------------------------------------------------
#define TBM 64
#define TBN 128
#define TBK 32
#define LDSTRIDE 40   // elements per LDS row (64B data + 16B pad)

__global__ __launch_bounds__(256)
void conv_wmma_k(const unsigned short* __restrict__ X,   // [N,Cin,H,W] bf16
                 const unsigned short* __restrict__ Wt,  // [Coutp,Ktotp] bf16 padded
                 const float* __restrict__ Bias,         // [Cout] f32
                 void* __restrict__ Y,                   // [N,Cout,Ho,Wo]
                 int N, int Cin, int H, int W, int Cout, int Ho, int Wo,
                 int kh, int kw, int stride, int pad, int dil,
                 int relu, int out_f32, int Ktot, int Ktotp, int Coutp)
{
    __shared__ __align__(16) unsigned short As [TBM][LDSTRIDE];
    __shared__ __align__(16) unsigned short BsT[TBN][LDSTRIDE];
    __shared__ int cOff[TBN], cIh[TBN], cIw[TBN];

    const int Ntot = N * Ho * Wo;
    const int khw  = kh * kw;
    const int n0   = blockIdx.x * TBN;
    const int m0   = blockIdx.y * TBM;
    const int tid  = threadIdx.x;
    const int lane = tid & 31;
    const int wv   = tid >> 5;
    const int wm   = (wv >> 2) * 32;   // wave M base (0/32)
    const int wn   = (wv & 3) * 32;    // wave N base (0/32/64/96)
    const int half = lane >> 4;
    const int l15  = lane & 15;

    // ---- per-column im2col precompute (K-invariant) ----
    if (tid < TBN) {
        int gn = n0 + tid;
        int offv = -1, ih0 = 0, iw0 = 0;
        if (gn < Ntot) {
            int b   = gn / (Ho * Wo);
            int rem = gn - b * Ho * Wo;
            int oh  = rem / Wo, ow = rem - oh * Wo;
            offv = b * Cin * H * W;
            ih0  = oh * stride - pad;
            iw0  = ow * stride - pad;
        }
        cOff[tid] = offv; cIh[tid] = ih0; cIw[tid] = iw0;
    }
    __syncthreads();

    const unsigned ldsA = (unsigned)(unsigned long long)(void*)&As[0][0];
    const int kk    = tid & 31;   // this thread's K column within the chunk
    const int nbase = tid >> 5;   // B columns nbase, nbase+8, ..., nbase+120

    v8f acc00 = {0,0,0,0,0,0,0,0}, acc01 = {0,0,0,0,0,0,0,0};
    v8f acc10 = {0,0,0,0,0,0,0,0}, acc11 = {0,0,0,0,0,0,0,0};

    for (int k0 = 0; k0 < Ktotp; k0 += TBK) {
        // ---- A tile via Tensor Data Mover (wave 0 issues the DMA) ----
        if (wv == 0) {
            tdm_load_a_tile(ldsA, Wt + (size_t)m0 * Ktotp + k0,
                            Ktotp - k0, Coutp - m0, Ktotp);
        }
        // ---- B tile: im2col, stored transposed [n][k] ----
        int gk = k0 + kk;
        int ci = 0, r = 0, s = 0;
        bool kok = gk < Ktot;
        if (kok) {
            ci = gk / khw;
            int r2 = gk - ci * khw;
            r = r2 / kw; s = r2 - r * kw;
        }
        #pragma unroll
        for (int j = 0; j < 16; ++j) {
            int nn = nbase + j * 8;
            int offv = cOff[nn];
            unsigned short v = 0;
            if (kok && offv >= 0) {
                int ih = cIh[nn] + r * dil;
                int iw = cIw[nn] + s * dil;
                if (ih >= 0 && ih < H && iw >= 0 && iw < W)
                    v = X[(size_t)offv + ((size_t)ci * H + ih) * W + iw];
            }
            BsT[nn][kk] = v;
        }
        if (wv == 0) __builtin_amdgcn_s_wait_tensorcnt(0);
        __syncthreads();

        // ---- fragments (128-bit LDS loads) + 4 WMMAs ----
        Frag32B fa0, fa1, fb0, fb1;
        const int rA0 = wm + l15, rA1 = wm + 16 + l15;
        const int cB0 = wn + l15, cB1 = wn + 16 + l15;
        fa0.lo = *(const uint4*)&As[rA0][half * 8];
        fa0.hi = *(const uint4*)&As[rA0][16 + half * 8];
        fa1.lo = *(const uint4*)&As[rA1][half * 8];
        fa1.hi = *(const uint4*)&As[rA1][16 + half * 8];
        fb0.lo = *(const uint4*)&BsT[cB0][half * 8];
        fb0.hi = *(const uint4*)&BsT[cB0][16 + half * 8];
        fb1.lo = *(const uint4*)&BsT[cB1][half * 8];
        fb1.hi = *(const uint4*)&BsT[cB1][16 + half * 8];
        v16bf a0 = __builtin_bit_cast(v16bf, fa0);
        v16bf a1 = __builtin_bit_cast(v16bf, fa1);
        v16bf b0 = __builtin_bit_cast(v16bf, fb0);
        v16bf b1 = __builtin_bit_cast(v16bf, fb1);

        acc00 = __builtin_amdgcn_wmma_f32_16x16x32_bf16(false, a0, false, b0, (short)0, acc00, false, false);
        acc01 = __builtin_amdgcn_wmma_f32_16x16x32_bf16(false, a0, false, b1, (short)0, acc01, false, false);
        acc10 = __builtin_amdgcn_wmma_f32_16x16x32_bf16(false, a1, false, b0, (short)0, acc10, false, false);
        acc11 = __builtin_amdgcn_wmma_f32_16x16x32_bf16(false, a1, false, b1, (short)0, acc11, false, false);
        __syncthreads();
    }

    // ---- epilogue: bias (+ReLU), scatter to NCHW ----
    auto store_tile = [&](const v8f& acc, int msub, int nsub) {
        #pragma unroll
        for (int v = 0; v < 8; ++v) {
            int mg = m0 + wm + msub + v + half * 8;
            int ng = n0 + wn + nsub + l15;
            if (mg < Cout && ng < Ntot) {
                float val = acc[v] + Bias[mg];
                if (relu) val = fmaxf(val, 0.f);
                int b   = ng / (Ho * Wo);
                int rem = ng - b * Ho * Wo;
                int oh  = rem / Wo, ow = rem - oh * Wo;
                size_t o = ((size_t)(b * Cout + mg) * Ho + oh) * Wo + ow;
                if (out_f32) ((float*)Y)[o] = val;
                else         ((unsigned short*)Y)[o] = f2bf(val);
            }
        }
    };
    store_tile(acc00,  0,  0);
    store_tile(acc01,  0, 16);
    store_tile(acc10, 16,  0);
    store_tile(acc11, 16, 16);
}

// ---------------------------------------------------------------------------
// Max pool (bf16). OOB / -inf padding handled by window clamping.
// ---------------------------------------------------------------------------
__global__ void maxpool_k(const unsigned short* __restrict__ X,
                          unsigned short* __restrict__ Y,
                          int N, int C, int H, int W, int Ho, int Wo,
                          int k, int s, int p)
{
    int i = blockIdx.x * blockDim.x + threadIdx.x;
    int tot = N * C * Ho * Wo;
    if (i >= tot) return;
    int ow = i % Wo; int t = i / Wo;
    int oh = t % Ho; t /= Ho;
    int c = t % C;  int n = t / C;
    float m = -3.402823466e38f;
    int h0 = oh * s - p, w0 = ow * s - p;
    for (int r = 0; r < k; ++r) {
        int ih = h0 + r; if (ih < 0 || ih >= H) continue;
        for (int q = 0; q < k; ++q) {
            int iw = w0 + q; if (iw < 0 || iw >= W) continue;
            m = fmaxf(m, bf2f(X[((n * C + c) * H + ih) * W + iw]));
        }
    }
    Y[i] = f2bf(m);
}

// ---------------------------------------------------------------------------
// L2-normalize f0 across channels; one wave32 per position, shfl_xor reduce.
// ---------------------------------------------------------------------------
__global__ __launch_bounds__(32)
void l2norm_k(unsigned short* __restrict__ F, const float* __restrict__ scale,
              int Npos, int C, int HW)
{
    int pos = blockIdx.x;
    if (pos >= Npos) return;
    int b  = pos / HW;
    int hw = pos - b * HW;
    int lane = threadIdx.x;
    float s = 0.f;
    for (int c = lane; c < C; c += 32) {
        float v = bf2f(F[(size_t)(b * C + c) * HW + hw]);
        s += v * v;
    }
    #pragma unroll
    for (int off = 16; off; off >>= 1) s += __shfl_xor(s, off, 32);
    float inv = 1.0f / fmaxf(sqrtf(s), 1e-12f);
    for (int c = lane; c < C; c += 32) {
        size_t idx = (size_t)(b * C + c) * HW + hw;
        F[idx] = f2bf(bf2f(F[idx]) * inv * scale[c]);
    }
}

// ---------------------------------------------------------------------------
// Prior boxes (one launch per level)
// ---------------------------------------------------------------------------
struct PriorLevel { float aw[6], ah[6]; int f, stride, A, offset; };

__global__ void priors_k(float* __restrict__ P, PriorLevel L)
{
    int i = blockIdx.x * blockDim.x + threadIdx.x;
    int tot = L.f * L.f * L.A;
    if (i >= tot) return;
    int a = i % L.A; int cell = i / L.A;
    int cx = cell % L.f, cy = cell / L.f;
    float* o = P + (size_t)(L.offset + i) * 4;
    o[0] = (cx + 0.5f) * (float)L.stride;
    o[1] = (cy + 0.5f) * (float)L.stride;
    o[2] = L.aw[a];
    o[3] = L.ah[a];
}

// ---------------------------------------------------------------------------
// Head permute: NCHW head-conv output -> gathered [B, totalAnch, T]
// ---------------------------------------------------------------------------
__global__ void head_gather_k(const float* __restrict__ src, float* __restrict__ dst,
                              int B, int Hd, int A, int T, int anchOff, int totalAnch)
{
    int i = blockIdx.x * blockDim.x + threadIdx.x;
    int tot = B * Hd * Hd * A * T;
    if (i >= tot) return;
    int t = i % T; int r = i / T;
    int a = r % A; r /= A;
    int w = r % Hd; r /= Hd;
    int h = r % Hd; int b = r / Hd;
    int anch = anchOff + (h * Hd + w) * A + a;
    dst[((size_t)b * totalAnch + anch) * T + t] =
        src[(((size_t)b * (A * T) + a * T + t) * Hd + h) * Hd + w];
}

// ---------------------------------------------------------------------------
// Softmax over 81 classes + SSD box decode. Output [B, 8732, 84]
// ---------------------------------------------------------------------------
__global__ void decode_k(const float* __restrict__ cls, const float* __restrict__ reg,
                         const float* __restrict__ priors, float* __restrict__ out,
                         int B, int NA)
{
    int i = blockIdx.x * blockDim.x + threadIdx.x;
    if (i >= B * NA) return;
    int a = i % NA;
    const float* L = cls + (size_t)i * 81;
    float m = L[0];
    for (int c = 1; c < 81; ++c) m = fmaxf(m, L[c]);
    float s = 0.f;
    for (int c = 0; c < 81; ++c) s += __expf(L[c] - m);
    float inv = 1.f / s;
    const float* R = reg + (size_t)i * 4;
    const float* P = priors + (size_t)a * 4;
    float pcx = P[0], pcy = P[1], pw = P[2], ph = P[3];
    float cx = R[0] * 0.1f * pw + pcx;
    float cy = R[1] * 0.1f * ph + pcy;
    const float CLMP = 4.135166556742356f; // log(1000/16)
    float w = pw * __expf(fminf(R[2] * 0.2f, CLMP));
    float h = ph * __expf(fminf(R[3] * 0.2f, CLMP));
    float* O = out + (size_t)i * 84;
    O[0] = fminf(fmaxf(cx - 0.5f * w, 0.f), 300.f);
    O[1] = fminf(fmaxf(cy - 0.5f * h, 0.f), 300.f);
    O[2] = fminf(fmaxf(cx + 0.5f * w, 0.f), 300.f);
    O[3] = fminf(fmaxf(cy + 0.5f * h, 0.f), 300.f);
    for (int c = 1; c < 81; ++c) O[3 + c] = __expf(L[c] - m) * inv;
}

// ---------------------------------------------------------------------------
// Host-side network description
// ---------------------------------------------------------------------------
struct CS { int ci, co, k, s, p, d; };
static const CS VGGS[15] = {
    {3,64,3,1,1,1},{64,64,3,1,1,1},{64,128,3,1,1,1},{128,128,3,1,1,1},
    {128,256,3,1,1,1},{256,256,3,1,1,1},{256,256,3,1,1,1},{256,512,3,1,1,1},
    {512,512,3,1,1,1},{512,512,3,1,1,1},{512,512,3,1,1,1},{512,512,3,1,1,1},
    {512,512,3,1,1,1},{512,1024,3,1,6,6},{1024,1024,1,1,0,1}};
static const CS EXTS[8] = {
    {1024,256,1,1,0,1},{256,512,3,2,1,1},{512,128,1,1,0,1},{128,256,3,2,1,1},
    {256,128,1,1,0,1},{128,256,3,1,0,1},{256,128,1,1,0,1},{128,256,3,1,0,1}};
static const int LCH[6]  = {512,1024,512,256,256,256};
static const int LNA[6]  = {4,6,6,6,4,4};
static const int LF[6]   = {38,19,10,5,3,1};
static const int LSTR[6] = {8,16,32,64,100,300};
static const int LOFF[6] = {0,5776,7942,8542,8692,8728};
static const int TOTAL_ANCH = 8732;

static inline int align32(int v) { return (v + 31) & ~31; }
static inline int align64(int v) { return (v + 63) & ~63; }

extern "C" void kernel_launch(void* const* d_in, const int* in_sizes, int n_in,
                              void* d_out, int out_size, void* d_ws, size_t ws_size,
                              hipStream_t stream)
{
    (void)in_sizes; (void)n_in; (void)out_size; (void)ws_size;
    const int B = 4;

    const float* x_f32 = (const float*)d_in[0];
    const void* const* vgg_w  = d_in + 1;    // 15
    const void* const* vgg_b  = d_in + 16;   // 15
    const void* const* ext_w  = d_in + 31;   // 8
    const void* const* ext_b  = d_in + 39;   // 8
    const void* const* cls_w  = d_in + 47;   // 6
    const void* const* cls_b  = d_in + 53;   // 6
    const void* const* reg_w  = d_in + 59;   // 6
    const void* const* reg_b  = d_in + 65;   // 6
    const float* l2s = (const float*)d_in[71];

    // ---- bump allocator over workspace ----
    char* ws = (char*)d_ws;
    size_t off = 0;
    auto alloc = [&](size_t bytes) -> void* {
        off = (off + 255) & ~(size_t)255;
        void* p = ws + off;
        off += bytes;
        return p;
    };

    const size_t ACT_ELEMS = (size_t)B * 64 * 300 * 300;
    unsigned short* sA = (unsigned short*)alloc(ACT_ELEMS * 2);
    unsigned short* sB = (unsigned short*)alloc(ACT_ELEMS * 2);
    unsigned short* f0 = (unsigned short*)alloc((size_t)B * 512 * 38 * 38 * 2);
    unsigned short* f1 = (unsigned short*)alloc((size_t)B * 1024 * 19 * 19 * 2);
    unsigned short* f2 = (unsigned short*)alloc((size_t)B * 512 * 10 * 10 * 2);
    unsigned short* f3 = (unsigned short*)alloc((size_t)B * 256 * 5 * 5 * 2);
    unsigned short* f4 = (unsigned short*)alloc((size_t)B * 256 * 3 * 3 * 2);
    unsigned short* f5 = (unsigned short*)alloc((size_t)B * 256 * 1 * 1 * 2);
    float* headscr = (float*)alloc((size_t)B * (4 * 81) * 38 * 38 * 4);
    float* clsbuf  = (float*)alloc((size_t)B * TOTAL_ANCH * 81 * 4);
    float* regbuf  = (float*)alloc((size_t)B * TOTAL_ANCH * 4 * 4);
    float* priorb  = (float*)alloc((size_t)TOTAL_ANCH * 4 * 4);

    // convert + pad weights to bf16 [Coutp, Ktotp]
    auto wpad = [&](const void* src, int Cout, int Ktot) -> unsigned short* {
        int Coutp = align64(Cout), Ktotp = align32(Ktot);
        size_t tot = (size_t)Coutp * Ktotp;
        unsigned short* d = (unsigned short*)alloc(tot * 2);
        w_cvt_pad_k<<<(unsigned)((tot + 255) / 256), 256, 0, stream>>>(
            (const float*)src, d, Cout, Ktot, Coutp, Ktotp);
        return d;
    };

    unsigned short* vggw[15];
    for (int i = 0; i < 15; ++i)
        vggw[i] = wpad(vgg_w[i], VGGS[i].co, VGGS[i].ci * VGGS[i].k * VGGS[i].k);
    unsigned short* extw[8];
    for (int i = 0; i < 8; ++i)
        extw[i] = wpad(ext_w[i], EXTS[i].co, EXTS[i].ci * EXTS[i].k * EXTS[i].k);
    unsigned short* clsw[6];
    unsigned short* regw[6];
    for (int i = 0; i < 6; ++i) {
        clsw[i] = wpad(cls_w[i], LNA[i] * 81, LCH[i] * 9);
        regw[i] = wpad(reg_w[i], LNA[i] * 4,  LCH[i] * 9);
    }

    // convert input into sA
    {
        size_t n = (size_t)B * 3 * 300 * 300;
        f32_to_bf16_k<<<(unsigned)((n + 255) / 256), 256, 0, stream>>>(x_f32, sA, (int)n);
    }

    // ---- layer runners ----
    int C = 3, Hd = 300, Wd = 300;
    unsigned short* cur = sA;

    auto conv = [&](const unsigned short* X, const unsigned short* Wt, const float* Bias,
                    void* Y, bool outf32, bool relu,
                    int Cin, int Hin, int Win, int Cout, int k, int s, int p, int d,
                    int& Ho, int& Wo) {
        Ho = (Hin + 2 * p - d * (k - 1) - 1) / s + 1;
        Wo = (Win + 2 * p - d * (k - 1) - 1) / s + 1;
        int Ktot = Cin * k * k;
        int Ktotp = align32(Ktot), Coutp = align64(Cout);
        dim3 grid((unsigned)((B * Ho * Wo + TBN - 1) / TBN),
                  (unsigned)((Cout + TBM - 1) / TBM));
        conv_wmma_k<<<grid, 256, 0, stream>>>(X, Wt, Bias, Y,
            B, Cin, Hin, Win, Cout, Ho, Wo, k, k, s, p, d,
            relu ? 1 : 0, outf32 ? 1 : 0, Ktot, Ktotp, Coutp);
    };

    auto pick = [&](unsigned short* want) -> unsigned short* {
        if (want) return want;
        return (cur == sA) ? sB : sA;
    };

    int vi = 0;
    auto cu = [&](unsigned short* dstOverride) {
        const CS& s = VGGS[vi];
        unsigned short* dst = pick(dstOverride);
        int Ho, Wo;
        conv(cur, vggw[vi], (const float*)vgg_b[vi], dst, false, true,
             C, Hd, Wd, s.co, s.k, s.s, s.p, s.d, Ho, Wo);
        cur = dst; C = s.co; Hd = Ho; Wd = Wo; ++vi;
    };
    auto pool = [&](int k, int s, int p, bool ceilm) {
        int Ho = ceilm ? ((Hd + 2 * p - k + s - 1) / s + 1) : ((Hd + 2 * p - k) / s + 1);
        int Wo = Ho;
        unsigned short* dst = pick(nullptr);
        int tot = B * C * Ho * Wo;
        maxpool_k<<<(unsigned)((tot + 255) / 256), 256, 0, stream>>>(
            cur, dst, B, C, Hd, Wd, Ho, Wo, k, s, p);
        cur = dst; Hd = Ho; Wd = Wo;
    };

    // ---- VGG backbone ----
    cu(nullptr); cu(nullptr); pool(2, 2, 0, false);              // 300->150
    cu(nullptr); cu(nullptr); pool(2, 2, 0, false);              // 150->75
    cu(nullptr); cu(nullptr); cu(nullptr); pool(2, 2, 0, true);  // 75->38 ceil
    cu(nullptr); cu(nullptr); cu(f0);                            // f0 @38
    pool(2, 2, 0, false);                                        // 38->19
    cu(nullptr); cu(nullptr); cu(nullptr); pool(3, 1, 1, false); // 19->19
    cu(nullptr); cu(f1);                                         // conv6(d6), conv7 -> f1

    // ---- extras ----
    unsigned short* ftgt[4] = {f2, f3, f4, f5};
    int ei = 0;
    for (int j = 0; j < 4; ++j) {
        for (int t = 0; t < 2; ++t) {
            const CS& s = EXTS[ei];
            unsigned short* dst = (t == 1) ? ftgt[j] : pick(nullptr);
            int Ho, Wo;
            conv(cur, extw[ei], (const float*)ext_b[ei], dst, false, true,
                 C, Hd, Wd, s.co, s.k, s.s, s.p, s.d, Ho, Wo);
            cur = dst; C = s.co; Hd = Ho; Wd = Wo; ++ei;
        }
    }

    // ---- L2-normalize f0 (in place) ----
    {
        int HW = 38 * 38;
        l2norm_k<<<(unsigned)(B * HW), 32, 0, stream>>>(f0, l2s, B * HW, 512, HW);
    }

    // ---- detection heads ----
    unsigned short* feats[6] = {f0, f1, f2, f3, f4, f5};
    for (int i = 0; i < 6; ++i) {
        int Hf = LF[i], Cf = LCH[i], A = LNA[i];
        int Ho, Wo;
        conv(feats[i], clsw[i], (const float*)cls_b[i], headscr, true, false,
             Cf, Hf, Hf, A * 81, 3, 1, 1, 1, Ho, Wo);
        {
            int tot = B * Hf * Hf * A * 81;
            head_gather_k<<<(unsigned)((tot + 255) / 256), 256, 0, stream>>>(
                headscr, clsbuf, B, Hf, A, 81, LOFF[i], TOTAL_ANCH);
        }
        conv(feats[i], regw[i], (const float*)reg_b[i], headscr, true, false,
             Cf, Hf, Hf, A * 4, 3, 1, 1, 1, Ho, Wo);
        {
            int tot = B * Hf * Hf * A * 4;
            head_gather_k<<<(unsigned)((tot + 255) / 256), 256, 0, stream>>>(
                headscr, regbuf, B, Hf, A, 4, LOFF[i], TOTAL_ANCH);
        }
    }

    // ---- priors ----
    {
        const double scales[7] = {0.07, 0.15, 0.33, 0.51, 0.69, 0.87, 1.05};
        for (int i = 0; i < 6; ++i) {
            double bs = scales[i] * 300.0;
            double extra = sqrt(scales[i] * scales[i + 1]) * 300.0;
            double rr[5]; int nr = 3;
            rr[0] = 1.0; rr[1] = 2.0; rr[2] = 0.5;
            if (i >= 1 && i <= 3) { rr[3] = 3.0; rr[4] = 1.0 / 3.0; nr = 5; }
            PriorLevel L;
            double aw[6], ah[6]; int na = 0;
            aw[na] = bs * sqrt(rr[0]); ah[na] = bs / sqrt(rr[0]); ++na;
            aw[na] = extra; ah[na] = extra; ++na;
            for (int q = 1; q < nr; ++q) {
                aw[na] = bs * sqrt(rr[q]); ah[na] = bs / sqrt(rr[q]); ++na;
            }
            for (int q = 0; q < na; ++q) {
                L.aw[q] = (float)(aw[q] < 300.0 ? aw[q] : 300.0);
                L.ah[q] = (float)(ah[q] < 300.0 ? ah[q] : 300.0);
            }
            for (int q = na; q < 6; ++q) { L.aw[q] = 0.f; L.ah[q] = 0.f; }
            L.f = LF[i]; L.stride = LSTR[i]; L.A = LNA[i]; L.offset = LOFF[i];
            int tot = L.f * L.f * L.A;
            priors_k<<<(unsigned)((tot + 255) / 256), 256, 0, stream>>>(priorb, L);
        }
    }

    // ---- softmax + box decode -> d_out [B, 8732, 84] ----
    {
        int tot = B * TOTAL_ANCH;
        decode_k<<<(unsigned)((tot + 255) / 256), 256, 0, stream>>>(
            clsbuf, regbuf, priorb, (float*)d_out, B, TOTAL_ANCH);
    }
}
